// RSSM_74045236183454
// MI455X (gfx1250) — matrix operand order repaired and checked by reading, not compiled, not run on hardware
//
#include <hip/hip_runtime.h>
#include <hip/hip_bf16.h>
#include <math.h>

typedef __attribute__((ext_vector_type(16))) _Float16 v16h;
typedef __attribute__((ext_vector_type(8)))  _Float16 v8h;
typedef __attribute__((ext_vector_type(8)))  float    v8f;

#define T_ 4
#define S_ 32
#define A_ 6
#define D_ 1024
#define H_ 1024
#define E_ 1536
#define B_ 64
#define L_ 64

// ---------------------------------------------------------------------------
// f32 (KxN) -> f16 transposed (NxK) weight conversion: WT[n][k] = W[k][n]
// Read side coalesced; one-time cost per launch, stays in L2.
// ---------------------------------------------------------------------------
__global__ __launch_bounds__(256)
void convert_transpose_kernel(const float* __restrict__ W, _Float16* __restrict__ WT,
                              int K, int N) {
    int idx = blockIdx.x * 256 + threadIdx.x;
    if (idx >= K * N) return;
    int k = idx / N, n = idx % N;
    WT[(size_t)n * K + k] = (_Float16)W[idx];
}

// ---------------------------------------------------------------------------
// One-time: deter0 = tanh(W_init); stoch0 = mean-head of img_out_mean(deter0)
// Single block, 256 threads, 4 columns per thread.
// ---------------------------------------------------------------------------
__global__ __launch_bounds__(256)
void stoch0_kernel(const float* __restrict__ W_init, const float* __restrict__ W_img,
                   const float* __restrict__ g, const float* __restrict__ bb,
                   const float* __restrict__ W_ims, const float* __restrict__ b_ims,
                   float* __restrict__ deter0_vec, float* __restrict__ stoch0_vec) {
    __shared__ float d0[D_];
    __shared__ float red[256], red2[256];
    __shared__ float xssh[H_];
    const int tid = threadIdx.x;
#pragma unroll
    for (int j = 0; j < 4; ++j) {
        int c = tid + j * 256;
        float v = tanhf(W_init[c]);
        d0[c] = v;
        deter0_vec[c] = v;
    }
    __syncthreads();
    // h = d0 @ W_img  (1 x 1024 x 1024 GEMV)
    float h[4] = {0.f, 0.f, 0.f, 0.f};
    for (int k = 0; k < D_; ++k) {
        float v = d0[k];
#pragma unroll
        for (int j = 0; j < 4; ++j) h[j] += v * W_img[(size_t)k * H_ + tid + j * 256];
    }
    float s = 0.f, s2 = 0.f;
#pragma unroll
    for (int j = 0; j < 4; ++j) { s += h[j]; s2 += h[j] * h[j]; }
    red[tid] = s; red2[tid] = s2; __syncthreads();
    for (int st = 128; st > 0; st >>= 1) {
        if (tid < st) { red[tid] += red[tid + st]; red2[tid] += red2[tid + st]; }
        __syncthreads();
    }
    float mean = red[0] * (1.f / H_);
    float var  = red2[0] * (1.f / H_) - mean * mean;
    float rstd = rsqrtf(var + 1e-5f);
    __syncthreads();
#pragma unroll
    for (int j = 0; j < 4; ++j) {
        int c = tid + j * 256;
        float hn = (h[j] - mean) * rstd * g[c] + bb[c];
        float mem = 0.f, xs = 0.f;
#pragma unroll
        for (int t = 0; t < T_; ++t) {
            mem += (hn - mem) * 0.5f;
            float sp = (mem >= 0.5f) ? 1.f : 0.f;
            mem *= (1.f - sp);
            xs += sp;
        }
        xssh[c] = xs * 0.25f;
    }
    __syncthreads();
    if (tid < S_) {
        float acc = b_ims[tid];
        for (int k = 0; k < H_; ++k) acc += xssh[k] * W_ims[k * (2 * S_) + tid];
        stoch0_vec[tid] = acc;
    }
}

// ---------------------------------------------------------------------------
// One-time: initialize carries from deter0 / stoch0
// ---------------------------------------------------------------------------
__global__ __launch_bounds__(256)
void init_carry_kernel(const float* __restrict__ d0, const float* __restrict__ s0,
                       float* __restrict__ deter_c, float* __restrict__ stoch_c) {
    int idx = blockIdx.x * 256 + threadIdx.x;
    if (idx < T_ * B_ * D_) deter_c[idx] = d0[idx & (D_ - 1)];
    int r = idx - T_ * B_ * D_;
    if (r >= 0 && r < B_ * S_) stoch_c[r] = s0[r & (S_ - 1)];
}

// ---------------------------------------------------------------------------
// Per step: blend prev deter with deter0 on is_first; emit f32 + f16 copies
// ---------------------------------------------------------------------------
__global__ __launch_bounds__(256)
void blend_dp_kernel(const float* __restrict__ deter_carry, const int* __restrict__ is_first,
                     const float* __restrict__ deter0_vec,
                     float* __restrict__ dpb, _Float16* __restrict__ dp16, int l) {
    int idx = blockIdx.x * 256 + threadIdx.x;   // T*B*D
    int d = idx & (D_ - 1);
    int b = (idx >> 10) & (B_ - 1);
    float f = (float)is_first[b * L_ + l];
    float v = deter_carry[idx] * (1.f - f) + deter0_vec[d] * f;
    dpb[idx]  = v;
    dp16[idx] = (_Float16)v;
}

// ---------------------------------------------------------------------------
// Per step: x = [blended stoch, scaled action]; h = LN(x @ W_inp); LIF -> SP f16
// One block per batch row, 256 threads, 4 columns each.
// ---------------------------------------------------------------------------
__global__ __launch_bounds__(256)
void inp_kernel(const float* __restrict__ stoch_carry, const float* __restrict__ stoch0_vec,
                const float* __restrict__ action, const int* __restrict__ is_first,
                const float* __restrict__ W_inp, const float* __restrict__ g,
                const float* __restrict__ bb, _Float16* __restrict__ SP, int l) {
    const int b = blockIdx.x, tid = threadIdx.x;
    __shared__ float xsh[40];
    __shared__ float red[256], red2[256];
    float f = (float)is_first[b * L_ + l];
    if (tid < S_) {
        xsh[tid] = stoch_carry[b * S_ + tid] * (1.f - f) + stoch0_vec[tid] * f;
    } else if (tid < S_ + A_) {
        int j = tid - S_;
        float a = action[(b * L_ + l) * A_ + j];
        float sc = 1.f / fmaxf(fabsf(a), 1.f);
        xsh[tid] = a * sc * (1.f - f);
    }
    __syncthreads();
    float h[4];
    float s = 0.f, s2 = 0.f;
#pragma unroll
    for (int j = 0; j < 4; ++j) {
        int col = tid + j * 256;
        float acc = 0.f;
#pragma unroll
        for (int k = 0; k < S_ + A_; ++k) acc += xsh[k] * W_inp[k * H_ + col];
        h[j] = acc; s += acc; s2 += acc * acc;
    }
    red[tid] = s; red2[tid] = s2; __syncthreads();
    for (int st = 128; st > 0; st >>= 1) {
        if (tid < st) { red[tid] += red[tid + st]; red2[tid] += red2[tid + st]; }
        __syncthreads();
    }
    float mean = red[0] * (1.f / H_);
    float var  = red2[0] * (1.f / H_) - mean * mean;
    float rstd = rsqrtf(var + 1e-5f);
    float hn[4], mem[4] = {0.f, 0.f, 0.f, 0.f};
#pragma unroll
    for (int j = 0; j < 4; ++j) {
        int col = tid + j * 256;
        hn[j] = (h[j] - mean) * rstd * g[col] + bb[col];
    }
#pragma unroll
    for (int t = 0; t < T_; ++t) {
#pragma unroll
        for (int j = 0; j < 4; ++j) {
            int col = tid + j * 256;
            mem[j] += (hn[j] - mem[j]) * 0.5f;
            float sp = (mem[j] >= 0.5f) ? 1.f : 0.f;
            mem[j] *= (1.f - sp);
            SP[(size_t)(t * B_ + b) * H_ + col] = (_Float16)sp;
        }
    }
}

// ---------------------------------------------------------------------------
// WMMA f16 GEMM: C(MxN,f32) = A(MxK,f16,row-major) @ B(KxN) given as
// BT = B^T (NxK, f16, row-major).  Block tile 64x128, K staged 32-deep in LDS.
// 8 waves laid out 4x2; each wave computes a 16x64 strip (4 WMMA tiles).
// Register-staged pipeline: global loads for panel i+1 overlap WMMA on panel i.
// All LDS traffic is 16-byte (b128) vector ops.
// Requires M%64==0, N%128==0, K%32==0.
// ---------------------------------------------------------------------------
#define BM 64
#define BN 128
#define BK 32

__global__ __launch_bounds__(256)
void gemm_f16_wmma(const _Float16* __restrict__ A, const _Float16* __restrict__ BT,
                   float* __restrict__ C, int M, int N, int K) {
    __shared__ __align__(16) _Float16 As[BM * BK];   // [m][k]
    __shared__ __align__(16) _Float16 Bs[BN * BK];   // [n][k]  (transposed tile)
    const int tid  = threadIdx.x;
    const int wave = tid >> 5, lane = tid & 31;
    const int bm = blockIdx.y * BM, bn = blockIdx.x * BN;
    const int wm = (wave >> 1) * 16;     // 0,16,32,48
    const int wn = (wave & 1) * 64;      // 0,64
    const int m_l = lane & 15, kh = lane >> 4;

    // cooperative staging assignments (16B per load)
    const int ar = tid >> 2, ac = (tid & 3) * 8;   // A: 64 rows x 32 k, 8 halves/thread
    const int br = tid >> 1, bc = (tid & 1) * 16;  // BT: 128 rows x 32 k, 16 halves/thread

    const _Float16* Ag = A  + (size_t)(bm + ar) * K + ac;
    const _Float16* Bg = BT + (size_t)(bn + br) * K + bc;

    // prologue: panel 0 into registers
    v8h a_reg  = *(const v8h*)Ag;
    v8h b_reg0 = *(const v8h*)Bg;
    v8h b_reg1 = *(const v8h*)(Bg + 8);

    v8f acc0 = {}, acc1 = {}, acc2 = {}, acc3 = {};

    for (int k0 = 0; k0 < K; k0 += BK) {
        // commit staged registers to LDS
        *(v8h*)(As + ar * BK + ac)     = a_reg;
        *(v8h*)(Bs + br * BK + bc)     = b_reg0;
        *(v8h*)(Bs + br * BK + bc + 8) = b_reg1;
        __syncthreads();

        // issue global loads for next panel (overlap with WMMA below)
        if (k0 + BK < K) {
            a_reg  = *(const v8h*)(Ag + k0 + BK);
            b_reg0 = *(const v8h*)(Bg + k0 + BK);
            b_reg1 = *(const v8h*)(Bg + k0 + BK + 8);
            if (k0 + 2 * BK < K) {  // prefetch panel i+2 -> global_prefetch_b8
                __builtin_prefetch(Ag + k0 + 2 * BK, 0, 1);
                __builtin_prefetch(Bg + k0 + 2 * BK, 0, 1);
            }
        }

        // A fragment: lanes 0-15 (M=lane) hold K {0..15}; lanes 16-31 {16..31}
        const _Float16* ap = As + (wm + m_l) * BK + 16 * kh;
        v8h alo = *(const v8h*)ap;
        v8h ahi = *(const v8h*)(ap + 8);
        v16h afrag = __builtin_shufflevector(alo, ahi,
            0, 1, 2, 3, 4, 5, 6, 7, 8, 9, 10, 11, 12, 13, 14, 15);

#pragma unroll
        for (int j = 0; j < 4; ++j) {
            // B fragment: lane n = wn+j*16+m_l needs K = e + 16*kh, e=0..15:
            // contiguous in the transposed tile -> two b128 LDS loads.
            const _Float16* bp = Bs + (wn + j * 16 + m_l) * BK + 16 * kh;
            v8h blo = *(const v8h*)bp;
            v8h bhi = *(const v8h*)(bp + 8);
            v16h bfrag = __builtin_shufflevector(blo, bhi,
                0, 1, 2, 3, 4, 5, 6, 7, 8, 9, 10, 11, 12, 13, 14, 15);
            v8f& acc = (j == 0) ? acc0 : (j == 1) ? acc1 : (j == 2) ? acc2 : acc3;
            acc = __builtin_amdgcn_wmma_f32_16x16x32_f16(false, afrag, false, bfrag,
                                                         (short)0, acc, false, false);
        }
        __syncthreads();
    }

#pragma unroll
    for (int j = 0; j < 4; ++j) {
        v8f acc = (j == 0) ? acc0 : (j == 1) ? acc1 : (j == 2) ? acc2 : acc3;
#pragma unroll
        for (int v = 0; v < 8; ++v) {
            int m = bm + wm + v + 8 * kh;       // C layout: VGPR v -> M = v (+8 for hi lanes)
            int n = bn + wn + j * 16 + m_l;
            C[(size_t)m * N + n] = acc[v];
        }
    }
}

// ---------------------------------------------------------------------------
// Per step: GRU gates (all T at once; dp = prev deter[t], independent across t)
// Writes deter carry (f32), obs A-matrix (f16), and d_out[..., 32:1056].
// ---------------------------------------------------------------------------
__global__ __launch_bounds__(256)
void gru_kernel(const float* __restrict__ GX, const float* __restrict__ GH,
                const float* __restrict__ b_gru, const float* __restrict__ dpb,
                float* __restrict__ deter_carry, _Float16* __restrict__ Aobs,
                float* __restrict__ out, int l) {
    int idx = blockIdx.x * 256 + threadIdx.x;   // T*B*D
    int d  = idx & (D_ - 1);
    int tb = idx >> 10;                          // t*B + b
    const size_t row = (size_t)tb * (3 * D_);
    float xr = GX[row + d]            + b_gru[d];
    float xz = GX[row + D_ + d]       + b_gru[D_ + d];
    float xn = GX[row + 2 * D_ + d]   + b_gru[2 * D_ + d];
    float hr = GH[row + d];
    float hz = GH[row + D_ + d];
    float hh = GH[row + 2 * D_ + d];
    float r = 1.f / (1.f + expf(-(xr + hr)));
    float z = 1.f / (1.f + expf(-(xz + hz)));
    float n = tanhf(xn + r * hh);
    float dp = dpb[idx];
    float dn = (1.f - z) * n + z * dp;
    deter_carry[idx] = dn;
    Aobs[(size_t)tb * (D_ + E_) + d] = (_Float16)dn;
    out[((size_t)tb * L_ + l) * (S_ + D_) + S_ + d] = dn;
}

// ---------------------------------------------------------------------------
// Per step: copy embed[:, :, l, :] (f32) into obs A-matrix columns D..D+E (f16)
// ---------------------------------------------------------------------------
__global__ __launch_bounds__(256)
void embed_conv_kernel(const float* __restrict__ embed, _Float16* __restrict__ Aobs, int l) {
    int idx = blockIdx.x * 256 + threadIdx.x;   // T*B*E
    int e  = idx % E_;
    int tb = idx / E_;
    Aobs[(size_t)tb * (D_ + E_) + D_ + e] = (_Float16)embed[((size_t)tb * L_ + l) * E_ + e];
}

// ---------------------------------------------------------------------------
// Per step: obs LN + LIF across T + posterior mean head.
// One block per batch row; mem carried across t in registers.
// ---------------------------------------------------------------------------
__global__ __launch_bounds__(256)
void obs_post_kernel(const float* __restrict__ Hobs, const float* __restrict__ g,
                     const float* __restrict__ bb, const float* __restrict__ W_obst,
                     const float* __restrict__ b_obst, float* __restrict__ stoch_carry,
                     float* __restrict__ out, int l) {
    const int b = blockIdx.x, tid = threadIdx.x;
    __shared__ float red[256], red2[256];
    __shared__ float xssh[H_];
    float mem[4] = {0.f, 0.f, 0.f, 0.f};
    float xs[4]  = {0.f, 0.f, 0.f, 0.f};
    for (int t = 0; t < T_; ++t) {
        const float* hrow = Hobs + (size_t)(t * B_ + b) * H_;
        float h[4];
        float s = 0.f, s2 = 0.f;
#pragma unroll
        for (int j = 0; j < 4; ++j) {
            h[j] = hrow[tid + j * 256];
            s += h[j]; s2 += h[j] * h[j];
        }
        red[tid] = s; red2[tid] = s2; __syncthreads();
        for (int st = 128; st > 0; st >>= 1) {
            if (tid < st) { red[tid] += red[tid + st]; red2[tid] += red2[tid + st]; }
            __syncthreads();
        }
        float mean = red[0] * (1.f / H_);
        float var  = red2[0] * (1.f / H_) - mean * mean;
        float rstd = rsqrtf(var + 1e-5f);
        __syncthreads();
#pragma unroll
        for (int j = 0; j < 4; ++j) {
            int col = tid + j * 256;
            float hn = (h[j] - mean) * rstd * g[col] + bb[col];
            mem[j] += (hn - mem[j]) * 0.5f;
            float sp = (mem[j] >= 0.5f) ? 1.f : 0.f;
            mem[j] *= (1.f - sp);
            xs[j] += sp;
        }
    }
#pragma unroll
    for (int j = 0; j < 4; ++j) xssh[tid + j * 256] = xs[j] * 0.25f;
    __syncthreads();
    if (tid < S_) {
        float acc = b_obst[tid];
        for (int k = 0; k < H_; ++k) acc += xssh[k] * W_obst[k * (2 * S_) + tid];
        stoch_carry[b * S_ + tid] = acc;
#pragma unroll
        for (int t = 0; t < T_; ++t)
            out[((size_t)(t * B_ + b) * L_ + l) * (S_ + D_) + tid] = acc;
    }
}

// ---------------------------------------------------------------------------
// Host driver
// ---------------------------------------------------------------------------
extern "C" void kernel_launch(void* const* d_in, const int* in_sizes, int n_in,
                              void* d_out, int out_size, void* d_ws, size_t ws_size,
                              hipStream_t stream) {
    (void)in_sizes; (void)n_in; (void)out_size; (void)ws_size;
    const float* embed    = (const float*)d_in[0];
    const float* action   = (const float*)d_in[1];
    const int*   is_first = (const int*)d_in[2];
    const float* W_init   = (const float*)d_in[3];
    const float* W_inp    = (const float*)d_in[4];
    const float* ln_inp_g = (const float*)d_in[5];
    const float* ln_inp_b = (const float*)d_in[6];
    const float* Wx_gru   = (const float*)d_in[7];
    const float* Wh_gru   = (const float*)d_in[8];
    const float* b_gru    = (const float*)d_in[9];
    const float* W_img    = (const float*)d_in[10];
    const float* ln_img_g = (const float*)d_in[11];
    const float* ln_img_b = (const float*)d_in[12];
    const float* W_obs    = (const float*)d_in[13];
    const float* ln_obs_g = (const float*)d_in[14];
    const float* ln_obs_b = (const float*)d_in[15];
    const float* W_ims    = (const float*)d_in[16];
    const float* b_ims    = (const float*)d_in[17];
    const float* W_obst   = (const float*)d_in[18];
    const float* b_obst   = (const float*)d_in[19];
    float* out = (float*)d_out;

    // workspace carve-out (all 256B aligned)
    char* w = (char*)d_ws;
    auto alloc = [&](size_t bytes) -> char* {
        char* p = w; w += (bytes + 255) & ~(size_t)255; return p;
    };
    _Float16* WXT16  = (_Float16*)alloc((size_t)D_ * 3 * D_ * 2);        // (3072 x 1024) = Wx^T
    _Float16* WHT16  = (_Float16*)alloc((size_t)D_ * 3 * D_ * 2);        // (3072 x 1024) = Wh^T
    _Float16* WOBST16= (_Float16*)alloc((size_t)(D_ + E_) * H_ * 2);     // (1024 x 2560) = Wobs^T
    _Float16* SP16   = (_Float16*)alloc((size_t)T_ * B_ * H_ * 2);       // 256x1024
    _Float16* DP16   = (_Float16*)alloc((size_t)T_ * B_ * D_ * 2);       // 256x1024
    float*    DPB    = (float*)   alloc((size_t)T_ * B_ * D_ * 4);       // blended prev deter
    float*    GX     = (float*)   alloc((size_t)T_ * B_ * 3 * D_ * 4);   // 256x3072
    float*    GH     = (float*)   alloc((size_t)T_ * B_ * 3 * D_ * 4);   // 256x3072
    _Float16* AOBS   = (_Float16*)alloc((size_t)T_ * B_ * (D_ + E_) * 2);// 256x2560
    float*    HOBS   = (float*)   alloc((size_t)T_ * B_ * H_ * 4);       // 256x1024
    float*    DETERC = (float*)   alloc((size_t)T_ * B_ * D_ * 4);       // carry
    float*    STOCHC = (float*)   alloc((size_t)B_ * S_ * 4);            // carry
    float*    D0VEC  = (float*)   alloc((size_t)D_ * 4);
    float*    S0VEC  = (float*)   alloc((size_t)S_ * 4);

    // ---- one-time precompute (idempotent, graph-safe) ----
    {
        int n1 = D_ * 3 * D_;
        convert_transpose_kernel<<<(n1 + 255) / 256, 256, 0, stream>>>(Wx_gru, WXT16, D_, 3 * D_);
        convert_transpose_kernel<<<(n1 + 255) / 256, 256, 0, stream>>>(Wh_gru, WHT16, D_, 3 * D_);
        int n2 = (D_ + E_) * H_;
        convert_transpose_kernel<<<(n2 + 255) / 256, 256, 0, stream>>>(W_obs, WOBST16, D_ + E_, H_);
        stoch0_kernel<<<1, 256, 0, stream>>>(W_init, W_img, ln_img_g, ln_img_b,
                                             W_ims, b_ims, D0VEC, S0VEC);
        int n3 = T_ * B_ * D_ + B_ * S_;
        init_carry_kernel<<<(n3 + 255) / 256, 256, 0, stream>>>(D0VEC, S0VEC, DETERC, STOCHC);
    }

    // ---- sequential scan over L steps ----
    const int M = T_ * B_;                 // 256
    for (int l = 0; l < L_; ++l) {
        blend_dp_kernel<<<(T_ * B_ * D_) / 256, 256, 0, stream>>>(
            DETERC, is_first, D0VEC, DPB, DP16, l);
        inp_kernel<<<B_, 256, 0, stream>>>(
            STOCHC, S0VEC, action, is_first, W_inp, ln_inp_g, ln_inp_b, SP16, l);
        gemm_f16_wmma<<<dim3((3 * D_) / BN, M / BM), 256, 0, stream>>>(
            SP16, WXT16, GX, M, 3 * D_, D_);
        gemm_f16_wmma<<<dim3((3 * D_) / BN, M / BM), 256, 0, stream>>>(
            DP16, WHT16, GH, M, 3 * D_, D_);
        gru_kernel<<<(T_ * B_ * D_) / 256, 256, 0, stream>>>(
            GX, GH, b_gru, DPB, DETERC, AOBS, out, l);
        embed_conv_kernel<<<(T_ * B_ * E_) / 256, 256, 0, stream>>>(embed, AOBS, l);
        gemm_f16_wmma<<<dim3(H_ / BN, M / BM), 256, 0, stream>>>(
            AOBS, WOBST16, HOBS, M, H_, D_ + E_);
        obs_post_kernel<<<B_, 256, 0, stream>>>(
            HOBS, ln_obs_g, ln_obs_b, W_obst, b_obst, STOCHC, out, l);
    }
}
